// Model_49572512531070
// MI455X (gfx1250) — compile-verified, matrix operand order checked by reading, not run on hardware
//
#include <hip/hip_runtime.h>
#include <hip/hip_bf16.h>
#include <cstdint>
#include <cstddef>

// ---------------------------------------------------------------------------
// Types for CDNA5 WMMA (wave32)
// ---------------------------------------------------------------------------
typedef __attribute__((ext_vector_type(16))) __bf16 v16bf;
typedef __attribute__((ext_vector_type(8)))  __bf16 v8bf;
typedef __attribute__((ext_vector_type(8)))  float  v8f;

#define N_NODES 8192
#define EDGES   262144
#define F_IN_D  512
#define H_DIM   256

__device__ __forceinline__ __bf16 f2bf(float f) {
  union { float f; uint32_t u; } v; v.f = f;
  uint32_t r = v.u + 0x7FFFu + ((v.u >> 16) & 1u);   // round-to-nearest-even
  uint16_t h = (uint16_t)(r >> 16);
  return __builtin_bit_cast(__bf16, h);
}

// Hardware fp32 atomic add (GLOBAL_ATOMIC_ADD_F32, non-returning, device scope).
// Avoids any CAS-loop expansion in the edge-scatter hot path.
__device__ __forceinline__ void atomic_add_f32_hw(float* addr, float val) {
  asm volatile("global_atomic_add_f32 %0, %1, off scope:SCOPE_DEV"
               :: "v"(addr), "v"(val) : "memory");
}

// ---------------------------------------------------------------------------
// Elementwise conversion kernels
// ---------------------------------------------------------------------------
__global__ void cvt_bf16_kernel(const float* __restrict__ in, __bf16* __restrict__ out, int n) {
  int i = blockIdx.x * blockDim.x + threadIdx.x;
  if (i < n) out[i] = f2bf(in[i]);
}

// in: [R,H,H] stored (k,h); out: [R,H,H] stored (h,k)  (transpose inner matrix)
__global__ void cvt_bf16_T_kernel(const float* __restrict__ in, __bf16* __restrict__ out, int R, int H) {
  int i = blockIdx.x * blockDim.x + threadIdx.x;
  int total = R * H * H;
  if (i >= total) return;
  int r = i / (H * H);
  int rem = i - r * H * H;
  int h = rem / H;
  int k = rem - h * H;
  out[i] = f2bf(in[r * H * H + k * H + h]);
}

// ---------------------------------------------------------------------------
// Degree kernels
// ---------------------------------------------------------------------------
__global__ void deg_count_kernel(const int* __restrict__ idx, float* __restrict__ deg, int E) {
  int i = blockIdx.x * blockDim.x + threadIdx.x;
  if (i < E) atomicAdd(&deg[idx[i]], 1.0f);
}

__global__ void rsqrt_clamp_kernel(float* __restrict__ deg, int n) {
  int i = blockIdx.x * blockDim.x + threadIdx.x;
  if (i < n) deg[i] = rsqrtf(fmaxf(deg[i], 1.0f));
}

// ---------------------------------------------------------------------------
// Row scale (x * rsqrt(deg_out)) -> bf16
// ---------------------------------------------------------------------------
__global__ void scale_rows_bf16_kernel(const float* __restrict__ x, const float* __restrict__ r,
                                       __bf16* __restrict__ out, int total, int H) {
  int i = blockIdx.x * blockDim.x + threadIdx.x;
  if (i >= total) return;
  int n = i / H;
  out[i] = f2bf(x[i] * r[n]);
}

// ---------------------------------------------------------------------------
// WMMA GEMM:  C[M,N] = A[M,K] @ B[N,K]^T (+bias[N]),  A/B bf16 row-major.
// One wave -> one 16x16 tile; 8 waves/block stacked along M.
// Fragment layouts per CDNA5 ISA 7.12.2 (16-bit A 16x32, C/D 16x16 f32).
// ---------------------------------------------------------------------------
__global__ void wmma_gemm_kernel(const __bf16* __restrict__ A,
                                 const __bf16* __restrict__ B,
                                 const float* __restrict__ bias,
                                 float* __restrict__ Cf,
                                 __bf16* __restrict__ Cb,
                                 int M, int N, int K) {
  int lane = threadIdx.x & 31;
  int wv   = threadIdx.x >> 5;
  int tm   = blockIdx.y * 8 + wv;     // 16-row tile index along M
  int tn   = blockIdx.x;              // 16-col tile index along N
  if (tm * 16 >= M) return;

  int row = lane & 15;                // A-row (M) / B-row (N) / C-col (N)
  int hi  = lane >> 4;                // K-half selector per ISA layout

  const __bf16* pA = A + (size_t)(tm * 16 + row) * K + hi * 8;
  const __bf16* pB = B + (size_t)(tn * 16 + row) * K + hi * 8;

  v8f acc = {};
  for (int k0 = 0; k0 < K; k0 += 32) {
    v8bf alo = *(const v8bf*)(pA + k0);
    v8bf ahi = *(const v8bf*)(pA + k0 + 16);
    v8bf blo = *(const v8bf*)(pB + k0);
    v8bf bhi = *(const v8bf*)(pB + k0 + 16);
    v16bf a, b;
#pragma unroll
    for (int j = 0; j < 8; ++j) {
      a[j] = alo[j]; a[8 + j] = ahi[j];
      b[j] = blo[j]; b[8 + j] = bhi[j];
    }
    acc = __builtin_amdgcn_wmma_f32_16x16x32_bf16(
        /*neg_a=*/false, a, /*neg_b=*/false, b,
        /*c_mod=*/(short)0, acc, /*reuse_a=*/false, /*reuse_b=*/false);
  }

  int cn = tn * 16 + row;
  float bv = bias ? bias[cn] : 0.0f;
#pragma unroll
  for (int r = 0; r < 8; ++r) {
    int cm = tm * 16 + hi * 8 + r;
    float v = acc[r] + bv;
    if (Cf) Cf[(size_t)cm * N + cn] = v;
    if (Cb) Cb[(size_t)cm * N + cn] = f2bf(v);
  }
}

// ---------------------------------------------------------------------------
// Fused edge scatter: out[d[e], :] += m[s[e], :] * rdeg_in[d[e]]
// One wave per edge (H=256 -> 8 f32 per lane), hardware f32 atomics.
// ---------------------------------------------------------------------------
__global__ void scatter_scaled_kernel(const float* __restrict__ m,
                                      const int* __restrict__ s, const int* __restrict__ d,
                                      const float* __restrict__ rd,
                                      float* __restrict__ out, int E) {
  int gid  = blockIdx.x * blockDim.x + threadIdx.x;
  int e    = gid >> 5;
  int lane = gid & 31;
  if (e >= E) return;
  int src = s[e], dst = d[e];
  float scale = rd[dst];
  const float* mp = m + (size_t)src * H_DIM;
  float*       op = out + (size_t)dst * H_DIM;
#pragma unroll
  for (int j = 0; j < 8; ++j)
    atomic_add_f32_hw(op + lane + j * 32, mp[lane + j * 32] * scale);
}

// out[n,k] = b0[k] + b1[k]  (bias init for the two relations landing here)
__global__ void init_bias2_kernel(float* __restrict__ out, const float* __restrict__ b0,
                                  const float* __restrict__ b1, int total) {
  int i = blockIdx.x * blockDim.x + threadIdx.x;
  if (i >= total) return;
  int k = i & (H_DIM - 1);
  out[i] = b0[k] + b1[k];
}

// ---------------------------------------------------------------------------
// BatchNorm (batch stats) + PReLU
// ---------------------------------------------------------------------------
__global__ void colstats_kernel(const float* __restrict__ x, float* __restrict__ sums,
                                float* __restrict__ sumsq) {
  int col = threadIdx.x;                  // 256 threads = 256 columns
  int r0  = blockIdx.x * 256;
  float s = 0.f, q = 0.f;
  for (int r = 0; r < 256; ++r) {
    float v = x[(size_t)(r0 + r) * H_DIM + col];
    s += v; q += v * v;
  }
  atomicAdd(&sums[col], s);
  atomicAdd(&sumsq[col], q);
}

__global__ void bn_finalize_kernel(const float* __restrict__ sums, const float* __restrict__ sumsq,
                                   float* __restrict__ mean, float* __restrict__ invstd, float invN) {
  int k = threadIdx.x;
  float mu  = sums[k] * invN;
  float var = sumsq[k] * invN - mu * mu;
  mean[k]   = mu;
  invstd[k] = rsqrtf(var + 1e-5f);
}

__global__ void bn_prelu_kernel(float* __restrict__ x, const float* __restrict__ mean,
                                const float* __restrict__ invstd, const float* __restrict__ g,
                                const float* __restrict__ b, const float* __restrict__ alpha,
                                int total) {
  int i = blockIdx.x * blockDim.x + threadIdx.x;
  if (i >= total) return;
  int k = i & (H_DIM - 1);
  float v = (x[i] - mean[k]) * invstd[k] * g[k] + b[k];
  float a = alpha[0];
  x[i] = v >= 0.f ? v : a * v;
}

// ---------------------------------------------------------------------------
// Semantic attention: per-node projected score -> global sum (one layer)
// ---------------------------------------------------------------------------
__global__ void att_score_kernel(const float* __restrict__ z, const float* __restrict__ W1,
                                 const float* __restrict__ b1, const float* __restrict__ w2,
                                 float* __restrict__ wsum) {
  __shared__ float zrow[H_DIM];
  __shared__ float red[128];
  int n = blockIdx.x;
  int t = threadIdx.x;                    // 128 threads
  zrow[t]       = z[(size_t)n * H_DIM + t];
  zrow[t + 128] = z[(size_t)n * H_DIM + t + 128];
  __syncthreads();
  float acc = b1[t];
  for (int k = 0; k < H_DIM; ++k) acc += zrow[k] * W1[t * H_DIM + k];
  red[t] = tanhf(acc) * w2[t];
  __syncthreads();
  for (int st = 64; st > 0; st >>= 1) {
    if (t < st) red[t] += red[t + st];
    __syncthreads();
  }
  if (t == 0) atomicAdd(wsum, red[0]);
}

__global__ void softmax3_kernel(const float* __restrict__ wsum, float* __restrict__ beta, float invN) {
  float w0 = wsum[0] * invN, w1 = wsum[1] * invN, w2 = wsum[2] * invN;
  float mx = fmaxf(w0, fmaxf(w1, w2));
  float e0 = expf(w0 - mx), e1 = expf(w1 - mx), e2 = expf(w2 - mx);
  float s = e0 + e1 + e2;
  beta[0] = e0 / s; beta[1] = e1 / s; beta[2] = e2 / s;
}

__global__ void combine3_kernel(const float* __restrict__ e0, const float* __restrict__ e1,
                                const float* __restrict__ e2, const float* __restrict__ beta,
                                __bf16* __restrict__ out, int n) {
  int i = blockIdx.x * blockDim.x + threadIdx.x;
  if (i >= n) return;
  out[i] = f2bf(beta[0] * e0[i] + beta[1] * e1[i] + beta[2] * e2[i]);
}

// ---------------------------------------------------------------------------
// Host side orchestration
// ---------------------------------------------------------------------------
extern "C" void kernel_launch(void* const* d_in, const int* in_sizes, int n_in,
                              void* d_out, int out_size, void* d_ws, size_t ws_size,
                              hipStream_t stream) {
  // ---- inputs (setup_inputs order) ----
  const float* x_drug  = (const float*)d_in[0];
  const float* x_dis   = (const float*)d_in[1];
  const int* eS[4] = {(const int*)d_in[2], (const int*)d_in[4], (const int*)d_in[6], (const int*)d_in[8]};
  const int* eD[4] = {(const int*)d_in[3], (const int*)d_in[5], (const int*)d_in[7], (const int*)d_in[9]};
  const float* W_drug = (const float*)d_in[10];
  const float* b_drug = (const float*)d_in[11];
  const float* W_dis  = (const float*)d_in[12];
  const float* b_dis  = (const float*)d_in[13];
  const float* gc_w[2] = {(const float*)d_in[14], (const float*)d_in[16]};
  const float* gc_b[2] = {(const float*)d_in[15], (const float*)d_in[17]};
  const float* bn_g[2] = {(const float*)d_in[18], (const float*)d_in[20]};
  const float* bn_b[2] = {(const float*)d_in[19], (const float*)d_in[21]};
  const float* prelu[2] = {(const float*)d_in[22], (const float*)d_in[23]};
  const float* aW1_dr = (const float*)d_in[24];
  const float* ab1_dr = (const float*)d_in[25];
  const float* aw2_dr = (const float*)d_in[26];
  const float* aW1_di = (const float*)d_in[27];
  const float* ab1_di = (const float*)d_in[28];
  const float* aw2_di = (const float*)d_in[29];
  const float* dec_WR = (const float*)d_in[30];
  const float* dec_WD = (const float*)d_in[31];
  float* out = (float*)d_out;

  // ---- workspace carve-out (256B aligned) ----
  size_t off = 0;
  auto alloc = [&](size_t bytes) -> void* {
    void* p = (char*)d_ws + off;
    off += (bytes + 255) & ~(size_t)255;
    return p;
  };
  const size_t NH  = (size_t)N_NODES * H_DIM;     // 2M elems
  const size_t NF  = (size_t)N_NODES * F_IN_D;    // 4M elems
  __bf16* xdr_bf  = (__bf16*)alloc(NF * 2);
  __bf16* xdi_bf  = (__bf16*)alloc(NF * 2);
  __bf16* Wdr_bf  = (__bf16*)alloc((size_t)H_DIM * F_IN_D * 2);
  __bf16* Wdi_bf  = (__bf16*)alloc((size_t)H_DIM * F_IN_D * 2);
  __bf16* gw_bf[2];
  gw_bf[0] = (__bf16*)alloc((size_t)4 * H_DIM * H_DIM * 2);
  gw_bf[1] = (__bf16*)alloc((size_t)4 * H_DIM * H_DIM * 2);
  __bf16* wR_bf   = (__bf16*)alloc((size_t)H_DIM * H_DIM * 2);
  __bf16* wD_bf   = (__bf16*)alloc((size_t)H_DIM * H_DIM * 2);
  float* rdeg     = (float*)alloc((size_t)8 * N_NODES * 4);   // rsqrt(clamped deg), 8 arrays
  float* emb_dr[3], *emb_di[3];
  for (int l = 0; l < 3; ++l) emb_dr[l] = (float*)alloc(NH * 4);
  for (int l = 0; l < 3; ++l) emb_di[l] = (float*)alloc(NH * 4);
  __bf16* scaled_bf = (__bf16*)alloc(NH * 2);
  float* m_f32      = (float*)alloc(NH * 4);
  float* sums   = (float*)alloc(H_DIM * 4);
  float* sumsq  = (float*)alloc(H_DIM * 4);
  float* meanv  = (float*)alloc(H_DIM * 4);
  float* invstd = (float*)alloc(H_DIM * 4);
  float* wsum_dr = (float*)alloc(64);
  float* beta_dr = (float*)alloc(64);
  float* wsum_di = (float*)alloc(64);
  float* beta_di = (float*)alloc(64);
  __bf16* hdr_bf = (__bf16*)alloc(NH * 2);
  __bf16* hdi_bf = (__bf16*)alloc(NH * 2);
  __bf16* R_bf   = (__bf16*)alloc(NH * 2);
  __bf16* D_bf   = (__bf16*)alloc(NH * 2);
  (void)ws_size; (void)n_in; (void)in_sizes; (void)out_size;

  auto grid1 = [](size_t n) { return dim3((unsigned)((n + 255) / 256)); };
  auto gemm = [&](const __bf16* A, const __bf16* B, const float* bias,
                  float* Cf, __bf16* Cb, int M, int N, int K) {
    dim3 g((unsigned)(N / 16), (unsigned)(M / 128));
    wmma_gemm_kernel<<<g, 256, 0, stream>>>(A, B, bias, Cf, Cb, M, N, K);
  };

  // ---- 1. degrees ----
  hipMemsetAsync(rdeg, 0, (size_t)8 * N_NODES * 4, stream);
  for (int r = 0; r < 4; ++r) {
    deg_count_kernel<<<grid1(EDGES), 256, 0, stream>>>(eS[r], rdeg + (size_t)(2 * r    ) * N_NODES, EDGES);
    deg_count_kernel<<<grid1(EDGES), 256, 0, stream>>>(eD[r], rdeg + (size_t)(2 * r + 1) * N_NODES, EDGES);
  }
  rsqrt_clamp_kernel<<<grid1(8 * N_NODES), 256, 0, stream>>>(rdeg, 8 * N_NODES);

  // ---- 2. bf16 conversions ----
  cvt_bf16_kernel<<<grid1(NF), 256, 0, stream>>>(x_drug, xdr_bf, (int)NF);
  cvt_bf16_kernel<<<grid1(NF), 256, 0, stream>>>(x_dis,  xdi_bf, (int)NF);
  cvt_bf16_kernel<<<grid1(H_DIM * F_IN_D), 256, 0, stream>>>(W_drug, Wdr_bf, H_DIM * F_IN_D);
  cvt_bf16_kernel<<<grid1(H_DIM * F_IN_D), 256, 0, stream>>>(W_dis,  Wdi_bf, H_DIM * F_IN_D);
  cvt_bf16_T_kernel<<<grid1(4 * H_DIM * H_DIM), 256, 0, stream>>>(gc_w[0], gw_bf[0], 4, H_DIM);
  cvt_bf16_T_kernel<<<grid1(4 * H_DIM * H_DIM), 256, 0, stream>>>(gc_w[1], gw_bf[1], 4, H_DIM);
  cvt_bf16_kernel<<<grid1(H_DIM * H_DIM), 256, 0, stream>>>(dec_WR, wR_bf, H_DIM * H_DIM);
  cvt_bf16_kernel<<<grid1(H_DIM * H_DIM), 256, 0, stream>>>(dec_WD, wD_bf, H_DIM * H_DIM);

  // ---- 3. input projections (WMMA) ----
  gemm(xdr_bf, Wdr_bf, b_drug, emb_dr[0], nullptr, N_NODES, H_DIM, F_IN_D);
  gemm(xdi_bf, Wdi_bf, b_dis,  emb_di[0], nullptr, N_NODES, H_DIM, F_IN_D);

  // ---- 4. two hetero GraphConv encoder layers ----
  // relation r: 0=dr->dr, 1=dr->di, 2=di->dr, 3=di->di
  // rdeg slots: [0]=drdr_s [1]=drdr_d [2]=drdi_s [3]=drdi_d [4]=didr_s [5]=didr_d [6]=didi_s [7]=didi_d
  for (int l = 0; l < 2; ++l) {
    const float* in_dr = emb_dr[l];
    const float* in_di = emb_di[l];
    float* out_dr = emb_dr[l + 1];
    float* out_di = emb_di[l + 1];

    struct Rel { const float* src; int rdeg_s, rdeg_d, w; const int *es, *ed; };
    // destination-grouped: [dst][incoming relation]
    Rel rels[2][2] = {
      { { in_dr, 0, 1, 0, eS[0], eD[0] },     // dr->dr (W0)
        { in_di, 4, 5, 2, eS[2], eD[2] } },   // di->dr (W2)
      { { in_dr, 2, 3, 1, eS[1], eD[1] },     // dr->di (W1)
        { in_di, 6, 7, 3, eS[3], eD[3] } },   // di->di (W3)
    };
    for (int t = 0; t < 2; ++t) {
      float* dst = t == 0 ? out_dr : out_di;
      // init with both relation biases, then atomically accumulate scaled messages
      init_bias2_kernel<<<grid1(NH), 256, 0, stream>>>(
          dst, gc_b[l] + (size_t)rels[t][0].w * H_DIM, gc_b[l] + (size_t)rels[t][1].w * H_DIM, (int)NH);
      for (int r = 0; r < 2; ++r) {
        const Rel& R = rels[t][r];
        scale_rows_bf16_kernel<<<grid1(NH), 256, 0, stream>>>(
            R.src, rdeg + (size_t)R.rdeg_s * N_NODES, scaled_bf, (int)NH, H_DIM);
        gemm(scaled_bf, gw_bf[l] + (size_t)R.w * H_DIM * H_DIM, nullptr,
             m_f32, nullptr, N_NODES, H_DIM, H_DIM);
        scatter_scaled_kernel<<<dim3(EDGES * 32 / 256), 256, 0, stream>>>(
            m_f32, R.es, R.ed, rdeg + (size_t)R.rdeg_d * N_NODES, dst, EDGES);
      }
      // BN + PReLU (shared gamma/beta, separate batch stats per tensor)
      hipMemsetAsync(sums, 0, H_DIM * 4, stream);
      hipMemsetAsync(sumsq, 0, H_DIM * 4, stream);
      colstats_kernel<<<dim3(N_NODES / 256), 256, 0, stream>>>(dst, sums, sumsq);
      bn_finalize_kernel<<<1, H_DIM, 0, stream>>>(sums, sumsq, meanv, invstd, 1.0f / N_NODES);
      bn_prelu_kernel<<<grid1(NH), 256, 0, stream>>>(dst, meanv, invstd, bn_g[l], bn_b[l], prelu[l], (int)NH);
    }
  }

  // ---- 5. semantic attention over the 3 layer embeddings ----
  hipMemsetAsync(wsum_dr, 0, 64, stream);
  hipMemsetAsync(wsum_di, 0, 64, stream);
  for (int l = 0; l < 3; ++l) {
    att_score_kernel<<<dim3(N_NODES), 128, 0, stream>>>(emb_dr[l], aW1_dr, ab1_dr, aw2_dr, wsum_dr + l);
    att_score_kernel<<<dim3(N_NODES), 128, 0, stream>>>(emb_di[l], aW1_di, ab1_di, aw2_di, wsum_di + l);
  }
  softmax3_kernel<<<1, 1, 0, stream>>>(wsum_dr, beta_dr, 1.0f / N_NODES);
  softmax3_kernel<<<1, 1, 0, stream>>>(wsum_di, beta_di, 1.0f / N_NODES);
  combine3_kernel<<<grid1(NH), 256, 0, stream>>>(emb_dr[0], emb_dr[1], emb_dr[2], beta_dr, hdr_bf, (int)NH);
  combine3_kernel<<<grid1(NH), 256, 0, stream>>>(emb_di[0], emb_di[1], emb_di[2], beta_di, hdi_bf, (int)NH);

  // ---- 6. decoder: R = h_drug @ WR^T, D = h_dis @ WD^T, out = R @ D^T ----
  gemm(hdr_bf, wR_bf, nullptr, nullptr, R_bf, N_NODES, H_DIM, H_DIM);
  gemm(hdi_bf, wD_bf, nullptr, nullptr, D_bf, N_NODES, H_DIM, H_DIM);
  gemm(R_bf, D_bf, nullptr, out, nullptr, N_NODES, N_NODES, H_DIM);
}
